// MHAttentionLayer_64587718197528
// MI455X (gfx1250) — matrix operand order repaired and checked by reading, not compiled, not run on hardware
//
#include <hip/hip_runtime.h>

// ---------------- bf16 helpers (raw ushort representation) ----------------
typedef __attribute__((ext_vector_type(16))) __bf16 v16bf;
typedef __attribute__((ext_vector_type(8)))  float  v8f;

union BFrag { v16bf v; unsigned u[8]; };

__device__ inline unsigned short f32_to_bf16(float x) {
    union { float f; unsigned u; } c; c.f = x;
    unsigned r = c.u + 0x7FFFu + ((c.u >> 16) & 1u);   // round-to-nearest-even
    return (unsigned short)(r >> 16);
}

__device__ inline v8f wmma_bf16(const BFrag& a, const BFrag& b, v8f c) {
    return __builtin_amdgcn_wmma_f32_16x16x32_bf16(
        /*neg_a=*/false, a.v, /*neg_b=*/false, b.v,
        /*c_mod=*/(short)0, c, /*reuse_a=*/false, /*reuse_b=*/false);
}

// ---------------- CDNA5 async global->LDS (ASYNCcnt-tracked DMA) ----------------
// Per-lane copy of 16 bytes: lds[vdst] = mem[vaddr]. GV mode (64-bit vaddr, saddr=off).
__device__ inline void async_copy_b128(unsigned lds_off, const void* gaddr) {
    asm volatile("global_load_async_to_lds_b128 %0, %1, off"
                 :: "v"(lds_off), "v"(gaddr) : "memory");
}
__device__ inline void wait_async_le3() { asm volatile("s_wait_asynccnt 0x3" ::: "memory"); }
__device__ inline void wait_async_0()   { asm volatile("s_wait_asynccnt 0x0" ::: "memory"); }

// ---------------- f32 -> bf16 convert + K-pad (zero fill) ----------------
__global__ __launch_bounds__(256) void cvt_pad_bf16(
    const float* __restrict__ in, unsigned short* __restrict__ outp,
    int R, int Kin, int Kpad)
{
    int idx = blockIdx.x * blockDim.x + threadIdx.x;   // one thread per 8 outputs
    int cpr = Kpad >> 3;                               // chunks per row
    if (idx >= R * cpr) return;
    int r  = idx / cpr;
    int c0 = (idx - r * cpr) << 3;
    unsigned w[4];
    #pragma unroll
    for (int j = 0; j < 4; ++j) {
        int c = c0 + 2 * j;
        unsigned lo = (c     < Kin) ? f32_to_bf16(in[(size_t)r * Kin + c])     : 0u;
        unsigned hi = (c + 1 < Kin) ? f32_to_bf16(in[(size_t)r * Kin + c + 1]) : 0u;
        w[j] = lo | (hi << 16);
    }
    uint4 v; v.x = w[0]; v.y = w[1]; v.z = w[2]; v.w = w[3];
    *(uint4*)&outp[(size_t)r * Kpad + c0] = v;
}

// ---------------- GEMM:  Out[M,N] = A[M,K] * W[N,K]^T + bias ----------------
// A, W raw-bf16 (K multiple of 32, zero-padded). bias f32. Out bf16 or f32.
// Block: 256 threads (8 waves), 128(M) x 64(N) C tile; each wave owns 32x32
// (2 A-frags x 2 B-frags -> 4 WMMAs per K-step). Double-buffered async staging.
template <bool OUT_BF16>
__global__ __launch_bounds__(256) void gemm_wmma(
    const unsigned short* __restrict__ Ab, const unsigned short* __restrict__ Wb,
    const float* __restrict__ bias, void* __restrict__ Out,
    int M, int N, int K, int tiles_n)
{
    // stride 72 elems = 144B: multiple of 16B (aligned b128 stores) and 36 dwords
    // per row -> 36*m mod 64 is a permutation over 16 consecutive rows
    __shared__ unsigned short lA[2][128][72];
    __shared__ unsigned short lW[2][64][72];

    const int blk = blockIdx.x;
    const int m0 = (blk / tiles_n) * 128;
    const int n0 = (blk % tiles_n) * 64;
    const int tid  = threadIdx.x;
    const int lane = tid & 31;
    const int wid  = tid >> 5;
    const int wm = wid & 3;          // m sub-tile: rows wm*32 .. wm*32+31
    const int wn = wid >> 2;         // n half:     cols wn*32 .. wn*32+31
    const int half = lane >> 4;
    const int l15  = lane & 15;

    // staging: per K-step, A tile 128x32 (512 x 16B chunks -> 2/thread),
    //          W tile 64x32 (256 chunks -> 1/thread)
    const int rA0 = tid >> 2;                 // chunk t     : rows 0..63
    const int rA1 = (tid + 256) >> 2;         // chunk t+256 : rows 64..127
    const int cS  = (tid & 3) * 8;            // col 0,8,16,24

    v8f acc00 = {}, acc01 = {}, acc10 = {}, acc11 = {};

    const int steps = K >> 5;                 // K multiple of 32

    // prologue: stage tile 0 into buffer 0
    {
        async_copy_b128((unsigned)(uintptr_t)&lA[0][rA0][cS], &Ab[(size_t)(m0 + rA0) * K + cS]);
        async_copy_b128((unsigned)(uintptr_t)&lA[0][rA1][cS], &Ab[(size_t)(m0 + rA1) * K + cS]);
        async_copy_b128((unsigned)(uintptr_t)&lW[0][rA0][cS], &Wb[(size_t)(n0 + rA0) * K + cS]);
    }

    for (int i = 0; i < steps; ++i) {
        const int ib = i & 1;
        if (i + 1 < steps) {
            const int kk = (i + 1) << 5;
            async_copy_b128((unsigned)(uintptr_t)&lA[ib ^ 1][rA0][cS],
                            &Ab[(size_t)(m0 + rA0) * K + kk + cS]);
            async_copy_b128((unsigned)(uintptr_t)&lA[ib ^ 1][rA1][cS],
                            &Ab[(size_t)(m0 + rA1) * K + kk + cS]);
            async_copy_b128((unsigned)(uintptr_t)&lW[ib ^ 1][rA0][cS],
                            &Wb[(size_t)(n0 + rA0) * K + kk + cS]);
            wait_async_le3();   // tile i's 3 copies (in-order completion) are done
        } else {
            wait_async_0();
        }
        __syncthreads();

        const unsigned short (*tA)[72] = lA[ib];
        const unsigned short (*tW)[72] = lW[ib];
        BFrag a0, a1, b0, b1;
        #pragma unroll
        for (int j = 0; j < 8; ++j) {
            int ka = ((j >> 2) * 16) + half * 8 + ((j & 3) * 2);       // A: 16x32
            a0.u[j] = *(const unsigned*)&tA[wm * 32 +      l15][ka];
            a1.u[j] = *(const unsigned*)&tA[wm * 32 + 16 + l15][ka];
            int kb = ((j >> 2) * 16) + half * 8 + ((j & 3) * 2);       // B: 32x16
            b0.u[j] = *(const unsigned*)&tW[wn * 32 +      l15][kb];
            b1.u[j] = *(const unsigned*)&tW[wn * 32 + 16 + l15][kb];
        }
        acc00 = wmma_bf16(a0, b0, acc00);
        acc01 = wmma_bf16(a0, b1, acc01);
        acc10 = wmma_bf16(a1, b0, acc10);
        acc11 = wmma_bf16(a1, b1, acc11);
        __syncthreads();    // protect buffer reuse before next issue
    }

    const int na = n0 + wn * 32 + l15;
    const int nb = na + 16;
    const float ba = bias[na];
    const float bb = bias[nb];
    #pragma unroll
    for (int i = 0; i < 8; ++i) {
        int mA = m0 + wm * 32 +      8 * half + i;   // C/D layout: m = 8*half + i
        int mB = m0 + wm * 32 + 16 + 8 * half + i;
        float x00 = acc00[i] + ba, x01 = acc01[i] + bb;
        float x10 = acc10[i] + ba, x11 = acc11[i] + bb;
        if (OUT_BF16) {
            unsigned short* O = (unsigned short*)Out;
            O[(size_t)mA * N + na] = f32_to_bf16(x00);
            O[(size_t)mA * N + nb] = f32_to_bf16(x01);
            O[(size_t)mB * N + na] = f32_to_bf16(x10);
            O[(size_t)mB * N + nb] = f32_to_bf16(x11);
        } else {
            float* O = (float*)Out;
            O[(size_t)mA * N + na] = x00;
            O[(size_t)mA * N + nb] = x01;
            O[(size_t)mB * N + na] = x10;
            O[(size_t)mB * N + nb] = x11;
        }
    }
}

// ---------------- Fused attention: scores -> softmax -> P*V ----------------
// Grid: B*H*(L/16) = 4096 blocks of 512 threads (16 waves).
// Q,K,V raw-bf16 [tokens, H*E=4096]; Ctx raw-bf16 [B*L, 4096].
__global__ __launch_bounds__(512) void attn_wmma(
    const unsigned short* __restrict__ Q,
    const unsigned short* __restrict__ Kp,
    const unsigned short* __restrict__ Vp,
    unsigned short* __restrict__ Ctx)
{
    __shared__ float sc[16][2048];   // 128 KB score/prob tile (within 320KB WGP LDS)

    const int blk = blockIdx.x;
    const int lt = blk & 63;           // L tile (L/16 = 64)
    const int h  = (blk >> 6) & 15;
    const int b  = blk >> 10;
    const int l0 = lt * 16;
    const int DHE = 4096;
    const int he  = h * 256;

    const int tid  = threadIdx.x;
    const int lane = tid & 31;
    const int wid  = tid >> 5;         // 0..15
    const int half = lane >> 4;
    const int l15  = lane & 15;

    // ---- preload Q A-fragments for all 8 K-steps over E=256 ----
    unsigned qf[8][8];
    {
        size_t qrow = ((size_t)(b * 1024 + l0 + l15)) * DHE + he;
        #pragma unroll
        for (int et = 0; et < 8; ++et) {
            #pragma unroll
            for (int i = 0; i < 8; ++i) {
                int ka = ((i >> 2) * 16) + half * 8 + ((i & 3) * 2);
                qf[et][i] = *(const unsigned*)&Q[qrow + et * 32 + ka];
            }
        }
    }

    // ---- phase 1: scores.  wave w covers s in [w*128, w*128+128) ----
    const float scale = 0.0625f;       // 1/sqrt(256)
    #pragma unroll
    for (int nt = 0; nt < 8; ++nt) {
        const int sb = wid * 128 + nt * 16;
        v8f acc = {};
        #pragma unroll
        for (int et = 0; et < 8; ++et) {
            BFrag a, bf;
            #pragma unroll
            for (int i = 0; i < 8; ++i) a.u[i] = qf[et][i];
            size_t krow = ((size_t)(sb + l15)) * DHE + he + et * 32;
            #pragma unroll
            for (int i = 0; i < 8; ++i) {
                int kb = ((i >> 2) * 16) + half * 8 + ((i & 3) * 2);
                bf.u[i] = *(const unsigned*)&Kp[krow + kb];
            }
            acc = wmma_bf16(a, bf, acc);
        }
        #pragma unroll
        for (int i = 0; i < 8; ++i)
            sc[8 * half + i][sb + l15] = acc[i] * scale;
    }
    __syncthreads();

    // ---- phase 2: softmax, one wave per L-row ----
    {
        const int row = wid;
        float mx = -3.0e38f;
        for (int j = lane; j < 2048; j += 32) mx = fmaxf(mx, sc[row][j]);
        #pragma unroll
        for (int off = 16; off; off >>= 1) mx = fmaxf(mx, __shfl_xor(mx, off, 32));
        float sum = 0.0f;
        for (int j = lane; j < 2048; j += 32) {
            float p = __expf(sc[row][j] - mx);
            sc[row][j] = p;
            sum += p;
        }
        #pragma unroll
        for (int off = 16; off; off >>= 1) sum += __shfl_xor(sum, off, 32);
        const float inv = 1.0f / sum;
        for (int j = lane; j < 2048; j += 32) sc[row][j] *= inv;
    }
    __syncthreads();

    // ---- phase 3: out[16,256] = P[16,2048] @ V[2048,256]; wave owns one E-tile ----
    {
        const int et = wid;                 // 16 tiles x 16 = 256
        const int e  = et * 16 + l15;       // B-frag: lane = n = e
        v8f acc = {};
        for (int st = 0; st < 64; ++st) {
            const int s0 = st * 32;
            BFrag a, bf;
            #pragma unroll
            for (int i = 0; i < 8; ++i) {
                int ka = ((i >> 2) * 16) + half * 8 + ((i & 3) * 2);
                float x0 = sc[l15][s0 + ka];
                float x1 = sc[l15][s0 + ka + 1];
                a.u[i] = (unsigned)f32_to_bf16(x0) | ((unsigned)f32_to_bf16(x1) << 16);
            }
            #pragma unroll
            for (int i = 0; i < 8; ++i) {
                int kb = ((i >> 2) * 16) + half * 8 + ((i & 3) * 2);   // k = s
                size_t v0 = ((size_t)(s0 + kb)) * DHE + he + e;
                unsigned lo = Vp[v0];
                unsigned hi = Vp[v0 + DHE];
                bf.u[i] = lo | (hi << 16);
            }
            acc = wmma_bf16(a, bf, acc);
        }
        #pragma unroll
        for (int i = 0; i < 8; ++i) {
            int m = 8 * half + i;
            Ctx[((size_t)(b * 1024 + l0 + m)) * DHE + he + e] = f32_to_bf16(acc[i]);
        }
    }
}

// ---------------- launcher ----------------
extern "C" void kernel_launch(void* const* d_in, const int* in_sizes, int n_in,
                              void* d_out, int out_size, void* d_ws, size_t ws_size,
                              hipStream_t stream) {
    (void)in_sizes; (void)n_in; (void)out_size; (void)ws_size;

    const float* tgt = (const float*)d_in[0];   // [4096,1024]
    const float* src = (const float*)d_in[1];   // [2048,1000]
    const float* val = (const float*)d_in[2];   // [2048,1000]
    const float* Wq  = (const float*)d_in[3];  const float* bq = (const float*)d_in[4];
    const float* Wk  = (const float*)d_in[5];  const float* bk = (const float*)d_in[6];
    const float* Wv  = (const float*)d_in[7];  const float* bv = (const float*)d_in[8];
    const float* Wo  = (const float*)d_in[9];  const float* bo = (const float*)d_in[10];
    float* out = (float*)d_out;                 // [4096,4096] f32

    // workspace layout (raw-bf16 elements)
    unsigned short* p   = (unsigned short*)d_ws;
    unsigned short* qb  = p;  p += (size_t)4096 * 4096;   // q
    unsigned short* kbf = p;  p += (size_t)2048 * 4096;   // k
    unsigned short* vbf = p;  p += (size_t)2048 * 4096;   // v
    unsigned short* cb  = p;  p += (size_t)4096 * 4096;   // ctx
    unsigned short* tb  = p;  p += (size_t)4096 * 1024;   // tgt bf16
    unsigned short* sb  = p;  p += (size_t)2048 * 1024;   // src bf16 (K padded)
    unsigned short* vlb = p;  p += (size_t)2048 * 1024;   // val bf16 (K padded)
    unsigned short* wqb = p;  p += (size_t)4096 * 1024;   // Wq bf16
    unsigned short* wkb = p;  p += (size_t)4096 * 1024;   // Wk bf16 (K padded)
    unsigned short* wvb = p;  p += (size_t)4096 * 1024;   // Wv bf16 (K padded)
    unsigned short* wob = p;                               // Wo bf16

    const dim3 blk(256);
    auto cvt = [&](const float* in, unsigned short* o, int R, int Kin, int Kpad) {
        int threads = R * (Kpad >> 3);
        cvt_pad_bf16<<<(threads + 255) / 256, blk, 0, stream>>>(in, o, R, Kin, Kpad);
    };
    // one-time bf16 conversion of all operands (bandwidth-bound, ~10us)
    cvt(tgt, tb,  4096, 1024, 1024);
    cvt(src, sb,  2048, 1000, 1024);
    cvt(val, vlb, 2048, 1000, 1024);
    cvt(Wq,  wqb, 4096, 1024, 1024);
    cvt(Wk,  wkb, 4096, 1000, 1024);
    cvt(Wv,  wvb, 4096, 1000, 1024);
    cvt(Wo,  wob, 4096, 4096, 4096);

    // projections (async-LDS double-buffered WMMA GEMMs, 128x64 tiles)
    gemm_wmma<true><<<32 * 64, blk, 0, stream>>>(tb,  wqb, bq, qb,  4096, 4096, 1024, 64);
    gemm_wmma<true><<<16 * 64, blk, 0, stream>>>(sb,  wkb, bk, kbf, 2048, 4096, 1024, 64);
    gemm_wmma<true><<<16 * 64, blk, 0, stream>>>(vlb, wvb, bv, vbf, 2048, 4096, 1024, 64);
    // attention: B*H*(L/16) blocks
    attn_wmma<<<4096, 512, 0, stream>>>(qb, kbf, vbf, cb);
    // output projection -> f32
    gemm_wmma<false><<<32 * 64, blk, 0, stream>>>(cb, wob, bo, out, 4096, 4096, 4096, 64);
}
